// Dlrm_62861141344492
// MI455X (gfx1250) — matrix operand order repaired and checked by reading, not compiled, not run on hardware
//
#include <hip/hip_runtime.h>
#include <hip/hip_bf16.h>

// ---------------------------------------------------------------------------
// DLRM forward for MI455X (gfx1250), wave32.
//  - GEMMs: v_wmma_f32_16x16x32_f16, 4 N-tiles register-blocked per wave.
//  - Embedding gather: global_load_async_to_lds_b128 (ASYNCcnt path).
//  - All inter-layer activations stay in LDS; weights f16 in L2-resident ws.
// ---------------------------------------------------------------------------

typedef __attribute__((ext_vector_type(16))) _Float16 v16h;
typedef __attribute__((ext_vector_type(8)))  float    v8f;

#define WMMA_F16(a, b, c) \
  __builtin_amdgcn_wmma_f32_16x16x32_f16(false, (a), false, (b), (short)0, (c), false, false)

// ---- Fragment loaders (contiguous K order, same for A and B) ----------------------
// A (16x32 MxK): lane<16 -> row M=lane, K0..15
//                lane>=16 -> row M=lane-16, K16..31
__device__ __forceinline__ v16h load_a_frag(const _Float16* base, int pitch,
                                            int m0, int kb) {
  const int lane = threadIdx.x & 31;
  const _Float16* p = base + (long)(m0 + (lane & 15)) * pitch + kb + ((lane >> 4) << 4);
  v16h f;
#pragma unroll
  for (int i = 0; i < 8; ++i) f[i] = p[i];
#pragma unroll
  for (int i = 0; i < 8; ++i) f[8 + i] = p[8 + i];
  return f;
}

// B (32x16 KxN) from row-major w[N][K]: lane<16 -> col N=lane, K=kb..kb+15;
// lane>=16 -> col N=lane-16, K=kb+16..kb+31 (contiguous 32B per lane).
__device__ __forceinline__ v16h load_b_frag(const _Float16* base, int pitch,
                                            int n0, int kb) {
  const int lane = threadIdx.x & 31;
  const _Float16* p = base + (long)(n0 + (lane & 15)) * pitch + kb + ((lane >> 4) << 4);
  v16h f;
#pragma unroll
  for (int i = 0; i < 16; ++i) f[i] = p[i];
  return f;
}

// f32-source variants (convert while building the fragment; cvt co-executes
// with WMMA which is tracked as a TRANS op on CDNA5).
__device__ __forceinline__ v16h load_a_frag_f32(const float* base, int pitch,
                                                int m0, int kb) {
  const int lane = threadIdx.x & 31;
  const float* p = base + (long)(m0 + (lane & 15)) * pitch + kb + ((lane >> 4) << 4);
  v16h f;
#pragma unroll
  for (int i = 0; i < 8; ++i) f[i] = (_Float16)p[i];
#pragma unroll
  for (int i = 0; i < 8; ++i) f[8 + i] = (_Float16)p[8 + i];
  return f;
}

__device__ __forceinline__ v16h load_b_frag_f32(const float* base, int pitch,
                                                int n0, int kb) {
  const int lane = threadIdx.x & 31;
  const float* p = base + (long)(n0 + (lane & 15)) * pitch + kb + ((lane >> 4) << 4);
  v16h f;
#pragma unroll
  for (int i = 0; i < 16; ++i) f[i] = (_Float16)p[i];
  return f;
}

// C/D layout: lane&15 = N within tile; vgpr r -> M = r + 8*(lane>=16).
__device__ __forceinline__ void store_acc(v8f acc, const float* __restrict__ bias,
                                          int n, _Float16* __restrict__ out,
                                          int out_pitch, int mb, bool relu) {
  const float bv = bias[n];
#pragma unroll
  for (int r = 0; r < 8; ++r) {
    float v = acc[r] + bv;
    if (relu) v = fmaxf(v, 0.0f);
    out[(long)(mb + r) * out_pitch + n] = (_Float16)v;
  }
}

// ---- Generic fused MLP layer: out = relu(in @ w.T + bias), f16 out --------
// Register-blocked: one A fragment feeds 4 B fragments / 4 accumulators.
// Requires N % 64 == 0 (holds for every layer here).
__device__ __forceinline__ void wmma_layer(
    const _Float16* __restrict__ in, int in_pitch,
    const _Float16* __restrict__ w, int w_pitch, int K,
    const float* __restrict__ bias,
    _Float16* __restrict__ out, int out_pitch,
    int Mrows, int N, bool relu)
{
  const int wave    = threadIdx.x >> 5;
  const int nwaves  = blockDim.x >> 5;
  const int lane    = threadIdx.x & 31;
  const int mtiles  = Mrows >> 4;
  const int ngroups = N >> 6;   // groups of 4 adjacent 16-wide N tiles
  for (int t = wave; t < mtiles * ngroups; t += nwaves) {
    const int mt = t / ngroups;
    const int n0 = (t % ngroups) << 6;
    __builtin_prefetch(w + (long)n0 * w_pitch, 0, 0);
    v8f acc0 = {}, acc1 = {}, acc2 = {}, acc3 = {};
    for (int kb = 0; kb < K; kb += 32) {
      v16h a  = load_a_frag(in, in_pitch, mt << 4, kb);
      v16h b0 = load_b_frag(w, w_pitch, n0,      kb);
      v16h b1 = load_b_frag(w, w_pitch, n0 + 16, kb);
      v16h b2 = load_b_frag(w, w_pitch, n0 + 32, kb);
      v16h b3 = load_b_frag(w, w_pitch, n0 + 48, kb);
      acc0 = WMMA_F16(a, b0, acc0);
      acc1 = WMMA_F16(a, b1, acc1);
      acc2 = WMMA_F16(a, b2, acc2);
      acc3 = WMMA_F16(a, b3, acc3);
    }
    const int nl = lane & 15;
    const int mb = (mt << 4) + ((lane >> 4) << 3);
    store_acc(acc0, bias, n0 + nl,      out, out_pitch, mb, relu);
    store_acc(acc1, bias, n0 + nl + 16, out, out_pitch, mb, relu);
    store_acc(acc2, bias, n0 + nl + 32, out, out_pitch, mb, relu);
    store_acc(acc3, bias, n0 + nl + 48, out, out_pitch, mb, relu);
  }
}

// ---- Kernel 0: f32 -> f16 weight conversion with K zero-padding -----------
__global__ void cvt_weights(const float* __restrict__ src, _Float16* __restrict__ dst,
                            int N, int K, int Kpad) {
  int i = blockIdx.x * blockDim.x + threadIdx.x;
  if (i >= N * Kpad) return;
  int n = i / Kpad, k = i % Kpad;
  dst[i] = (k < K) ? (_Float16)src[(long)n * K + k] : (_Float16)0.0f;
}

// ---- Kernel 1: fused bottom MLP 13->512->256->128 (64 rows / block) -------
__global__ void __launch_bounds__(256) bottom_mlp_kernel(
    const float* __restrict__ num,
    const _Float16* __restrict__ bw0h, const float* __restrict__ bb0,
    const _Float16* __restrict__ bw1h, const float* __restrict__ bb1,
    const _Float16* __restrict__ bw2h, const float* __restrict__ bb2,
    _Float16* __restrict__ bottom_h)
{
  __shared__ _Float16 xa[64 * 40];    // 13 padded to 32, pitch 40
  __shared__ _Float16 h0[64 * 520];   // 512, pitch 520
  __shared__ _Float16 h1[64 * 264];   // 256, pitch 264
  const int row0 = blockIdx.x * 64;
  for (int idx = threadIdx.x; idx < 64 * 32; idx += blockDim.x) {
    int r = idx >> 5, c = idx & 31;
    xa[r * 40 + c] = (c < 13) ? (_Float16)num[(long)(row0 + r) * 13 + c] : (_Float16)0.0f;
  }
  __syncthreads();
  wmma_layer(xa, 40, bw0h, 32, 32, bb0, h0, 520, 64, 512, true);
  __syncthreads();
  wmma_layer(h0, 520, bw1h, 512, 512, bb1, h1, 264, 64, 256, true);
  __syncthreads();
  wmma_layer(h1, 264, bw2h, 256, 256, bb2, bottom_h + (long)row0 * 128, 128, 64, 128, true);
}

// ---- Kernel 2: async embedding gather + WMMA self-interaction -------------
#define VPITCH 132   // f32 pitch: 128 + 4 (keeps each lane's 16B chunk aligned)
__global__ void __launch_bounds__(256) interact_kernel(
    const _Float16* __restrict__ bottom_h,
    const int* __restrict__ cat,
    const float* __restrict__ emb,
    _Float16* __restrict__ topin)
{
  __shared__ float Vf[8][32 * VPITCH];  // 27 rows (pad 32) x 128 f32
  __shared__ float G[8][32 * 33];       // Gram scratch per wave
  const int wave = threadIdx.x >> 5;
  const int lane = threadIdx.x & 31;
  const long s = (long)blockIdx.x * 8 + wave;
  float* Vw = Vf[wave];
  float* Gw = G[wave];

  // rows 1..26: async DMA straight into LDS, 16B per lane per row (512B/row).
  for (int t = 0; t < 26; ++t) {
    const int idx = cat[s * 26 + t];
    const float* e = emb + ((long)t * 100000 + (long)idx) * 128 + lane * 4;
    unsigned ldsoff = (unsigned)(unsigned long long)(const void*)
                      &Vw[(t + 1) * VPITCH + lane * 4];
    unsigned long long ga = (unsigned long long)e;
    asm volatile("global_load_async_to_lds_b128 %0, %1, off"
                 :: "v"(ldsoff), "v"(ga) : "memory");
  }
  // row 0: bottom MLP output (f16 -> f32)
  {
    const _Float16* src = bottom_h + s * 128;
    for (int i = lane; i < 128; i += 32) Vw[i] = (float)src[i];
  }
  // rows 27..31 zero
  for (int r = 27; r < 32; ++r)
    for (int i = lane; i < 128; i += 32) Vw[r * VPITCH + i] = 0.0f;
  asm volatile("s_wait_asynccnt 0x0" ::: "memory");
  __syncthreads();

  // Gram = V @ V^T, 32x32 via 2x2 tiles x 4 K-steps (16 WMMAs/sample)
  v8f a00 = {}, a01 = {}, a10 = {}, a11 = {};
  for (int kb = 0; kb < 128; kb += 32) {
    v16h fa0 = load_a_frag_f32(Vw, VPITCH, 0,  kb);
    v16h fa1 = load_a_frag_f32(Vw, VPITCH, 16, kb);
    v16h fb0 = load_b_frag_f32(Vw, VPITCH, 0,  kb);
    v16h fb1 = load_b_frag_f32(Vw, VPITCH, 16, kb);
    a00 = WMMA_F16(fa0, fb0, a00);
    a01 = WMMA_F16(fa0, fb1, a01);
    a10 = WMMA_F16(fa1, fb0, a10);
    a11 = WMMA_F16(fa1, fb1, a11);
  }
  {
    const int n  = lane & 15;
    const int mh = (lane >> 4) << 3;
#pragma unroll
    for (int r = 0; r < 8; ++r) {
      Gw[(mh + r)      * 33 + n]      = a00[r];
      Gw[(mh + r)      * 33 + n + 16] = a01[r];
      Gw[(mh + r + 16) * 33 + n]      = a10[r];
      Gw[(mh + r + 16) * 33 + n + 16] = a11[r];
    }
  }
  __syncthreads();

  // top-MLP input: [bottom(128) | tril(351) | zero pad to 512]
  _Float16* out = topin + s * 512;
  for (int i = lane; i < 128; i += 32) out[i] = (_Float16)Vw[i];
  for (int e = lane; e < 351; e += 32) {
    int i = (int)((1.0f + sqrtf(1.0f + 8.0f * (float)e)) * 0.5f);
    while (i * (i - 1) / 2 > e) --i;
    while ((i + 1) * i / 2 <= e) ++i;
    int j = e - i * (i - 1) / 2;
    out[128 + e] = (_Float16)Gw[i * 33 + j];
  }
  for (int i = 479 + lane; i < 512; i += 32) out[i] = (_Float16)0.0f;
}

// ---- Kernel 3: fused top MLP 480(512)->1024->1024->512->256->1 ------------
__global__ void __launch_bounds__(256) top_mlp_kernel(
    const _Float16* __restrict__ topin,
    const _Float16* __restrict__ tw0h, const float* __restrict__ tb0,
    const _Float16* __restrict__ tw1h, const float* __restrict__ tb1,
    const _Float16* __restrict__ tw2h, const float* __restrict__ tb2,
    const _Float16* __restrict__ tw3h, const float* __restrict__ tb3,
    const float* __restrict__ tw4, const float* __restrict__ tb4,
    float* __restrict__ out)
{
  __shared__ _Float16 bufA[32 * 1032];  // ping
  __shared__ _Float16 bufB[32 * 1032];  // pong
  const int row0 = blockIdx.x * 32;
  for (int idx = threadIdx.x; idx < 32 * 512; idx += blockDim.x) {
    int r = idx >> 9, c = idx & 511;
    bufA[r * 1032 + c] = topin[(long)(row0 + r) * 512 + c];
  }
  __syncthreads();
  wmma_layer(bufA, 1032, tw0h, 512,  512,  tb0, bufB, 1032, 32, 1024, true);
  __syncthreads();
  wmma_layer(bufB, 1032, tw1h, 1024, 1024, tb1, bufA, 1032, 32, 1024, true);
  __syncthreads();
  wmma_layer(bufA, 1032, tw2h, 1024, 1024, tb2, bufB, 1032, 32, 512, true);
  __syncthreads();
  wmma_layer(bufB, 1032, tw3h, 512,  512,  tb3, bufA, 1032, 32, 256, true);
  __syncthreads();
  if (threadIdx.x < 32) {
    float acc = tb4[0];
    const _Float16* rowp = bufA + threadIdx.x * 1032;
    for (int k = 0; k < 256; ++k) acc += (float)rowp[k] * tw4[k];
    out[row0 + threadIdx.x] = acc;
  }
}

// ---- Workspace layout (in f16 elements) -----------------------------------
#define OFF_BW0    0L                       // 512 x 32
#define OFF_BW1    (OFF_BW0 + 512L * 32)    // 256 x 512
#define OFF_BW2    (OFF_BW1 + 256L * 512)   // 128 x 256
#define OFF_TW0    (OFF_BW2 + 128L * 256)   // 1024 x 512
#define OFF_TW1    (OFF_TW0 + 1024L * 512)  // 1024 x 1024
#define OFF_TW2    (OFF_TW1 + 1024L * 1024) // 512 x 1024
#define OFF_TW3    (OFF_TW2 + 512L * 1024)  // 256 x 512
#define OFF_BOT    (OFF_TW3 + 256L * 512)   // 16384 x 128
#define OFF_TOPIN  (OFF_BOT + 16384L * 128) // 16384 x 512

extern "C" void kernel_launch(void* const* d_in, const int* in_sizes, int n_in,
                              void* d_out, int out_size, void* d_ws, size_t ws_size,
                              hipStream_t stream) {
  const float* num = (const float*)d_in[0];
  const int*   cat = (const int*)d_in[1];
  const float* emb = (const float*)d_in[2];
  const float* bw0 = (const float*)d_in[3];  const float* bb0 = (const float*)d_in[4];
  const float* bw1 = (const float*)d_in[5];  const float* bb1 = (const float*)d_in[6];
  const float* bw2 = (const float*)d_in[7];  const float* bb2 = (const float*)d_in[8];
  const float* tw0 = (const float*)d_in[9];  const float* tb0 = (const float*)d_in[10];
  const float* tw1 = (const float*)d_in[11]; const float* tb1 = (const float*)d_in[12];
  const float* tw2 = (const float*)d_in[13]; const float* tb2 = (const float*)d_in[14];
  const float* tw3 = (const float*)d_in[15]; const float* tb3 = (const float*)d_in[16];
  const float* tw4 = (const float*)d_in[17]; const float* tb4 = (const float*)d_in[18];
  float* out = (float*)d_out;

  _Float16* ws    = (_Float16*)d_ws;
  _Float16* bw0h  = ws + OFF_BW0;
  _Float16* bw1h  = ws + OFF_BW1;
  _Float16* bw2h  = ws + OFF_BW2;
  _Float16* tw0h  = ws + OFF_TW0;
  _Float16* tw1h  = ws + OFF_TW1;
  _Float16* tw2h  = ws + OFF_TW2;
  _Float16* tw3h  = ws + OFF_TW3;
  _Float16* both  = ws + OFF_BOT;
  _Float16* topin = ws + OFF_TOPIN;

  auto cvt = [&](const float* s, _Float16* d, int N, int K, int Kpad) {
    int total = N * Kpad;
    cvt_weights<<<(total + 255) / 256, 256, 0, stream>>>(s, d, N, K, Kpad);
  };
  cvt(bw0, bw0h, 512, 13, 32);
  cvt(bw1, bw1h, 256, 512, 512);
  cvt(bw2, bw2h, 128, 256, 256);
  cvt(tw0, tw0h, 1024, 480, 512);
  cvt(tw1, tw1h, 1024, 1024, 1024);
  cvt(tw2, tw2h, 512, 1024, 1024);
  cvt(tw3, tw3h, 256, 512, 512);

  bottom_mlp_kernel<<<16384 / 64, 256, 0, stream>>>(num, bw0h, bb0, bw1h, bb1,
                                                    bw2h, bb2, both);
  interact_kernel<<<16384 / 8, 256, 0, stream>>>(both, cat, emb, topin);
  top_mlp_kernel<<<16384 / 32, 256, 0, stream>>>(topin, tw0h, tb0, tw1h, tb1,
                                                 tw2h, tb2, tw3h, tb3, tw4, tb4, out);
}
